// mini_eMLP_74208444940655
// MI455X (gfx1250) — matrix operand order repaired and checked by reading, not compiled, not run on hardware
//
#include <hip/hip_runtime.h>
#include <stdint.h>

// Problem constants (match reference)
#define B_   128
#define T_   200
#define IN_  2312
#define H_   512
#define WWIN 10
#define OUTW 191   // T - W + 1

typedef __attribute__((ext_vector_type(16))) __bf16 v16bf;
typedef __attribute__((ext_vector_type(8)))  __bf16 v8bf;
typedef __attribute__((ext_vector_type(8)))  float  v8f;

// Native conversion: lowers to v_cvt(_pk)_bf16_f32 (RNE) instead of a 4-op
// integer bit-trick per element.
static __device__ __forceinline__ __bf16 f2bf(float f) { return (__bf16)f; }
static __device__ __forceinline__ __bf16 bfzero() {
  uint16_t z = 0;
  return __builtin_bit_cast(__bf16, z);
}
// 16 contiguous bf16 (16B-aligned) -> v16bf via two 16B loads
static __device__ __forceinline__ v16bf ld16bf(const uint16_t* p) {
  const v8bf* q = reinterpret_cast<const v8bf*>(p);
  return __builtin_shufflevector(q[0], q[1],
      0, 1, 2, 3, 4, 5, 6, 7, 8, 9, 10, 11, 12, 13, 14, 15);
}

// ---------------------------------------------------------------------------
// fp32 -> bf16 weight conversion
// ---------------------------------------------------------------------------
__global__ __launch_bounds__(256) void cvt_bf16_kernel(
    const float* __restrict__ src, uint16_t* __restrict__ dst, int n) {
  int i = blockIdx.x * 256 + threadIdx.x;
  if (i < n) dst[i] = __builtin_bit_cast(uint16_t, f2bf(src[i]));
}

// ---------------------------------------------------------------------------
// Phase 1: i2h[m, n] = sum_k input[m, k] * Wi[n, k]    (m = b*T + t)
// Block: 256 threads = 8 waves; block tile M128 x N64; wave tile M16 x N64
// (A fragment reused across 4 WMMAs per k-step).
// ---------------------------------------------------------------------------
__global__ __launch_bounds__(256) void i2h_gemm_kernel(
    const float* __restrict__ inp, const uint16_t* __restrict__ wibf,
    float* __restrict__ i2h) {
  const int lane   = threadIdx.x & 31;
  const int wave   = threadIdx.x >> 5;
  const int laneLo = lane & 15;
  const int laneHi = (lane >> 4) & 1;
  const int nb = blockIdx.x & 7;    // 8 column blocks of 64
  const int mb = blockIdx.x >> 3;   // 200 row blocks of 128
  const int m0 = mb * 128 + wave * 16;
  const int n0 = nb * 64;

  const float*    Arow = inp  + (size_t)(m0 + laneLo) * IN_;
  const uint16_t* Br0  = wibf + (size_t)(n0 +  0 + laneLo) * IN_;
  const uint16_t* Br1  = wibf + (size_t)(n0 + 16 + laneLo) * IN_;
  const uint16_t* Br2  = wibf + (size_t)(n0 + 32 + laneLo) * IN_;
  const uint16_t* Br3  = wibf + (size_t)(n0 + 48 + laneLo) * IN_;

  v8f c0 = {}, c1 = {}, c2 = {}, c3 = {};

  int k0 = 0;
  for (; k0 + 32 <= IN_; k0 += 32) {             // 72 full K-steps
    const int kbA = k0 + laneHi * 16;
    const int kbB = k0 + laneHi * 16;
    v16bf a;
#pragma unroll
    for (int j = 0; j < 8; ++j) {                // A: K = {kbA..+15}
      a[j]     = f2bf(Arow[kbA + j]);
      a[j + 8] = f2bf(Arow[kbA + 8 + j]);
    }
    v16bf b0v = ld16bf(Br0 + kbB);
    v16bf b1v = ld16bf(Br1 + kbB);
    v16bf b2v = ld16bf(Br2 + kbB);
    v16bf b3v = ld16bf(Br3 + kbB);
    c0 = __builtin_amdgcn_wmma_f32_16x16x32_bf16(false, a, false, b0v, (short)0, c0, false, false);
    c1 = __builtin_amdgcn_wmma_f32_16x16x32_bf16(false, a, false, b1v, (short)0, c1, false, false);
    c2 = __builtin_amdgcn_wmma_f32_16x16x32_bf16(false, a, false, b2v, (short)0, c2, false, false);
    c3 = __builtin_amdgcn_wmma_f32_16x16x32_bf16(false, a, false, b3v, (short)0, c3, false, false);
  }
  {                                              // K tail (8 valid), zero-padded
    const int kbA = k0 + laneHi * 16;
    const int kbB = k0 + laneHi * 16;
    v16bf a, b0v, b1v, b2v, b3v;
#pragma unroll
    for (int j = 0; j < 16; ++j) {
      const int kA = kbA + j;
      const int kB = kbB + j;
      const bool okA = (kA < IN_);
      const bool okB = (kB < IN_);
      // Branch-free: unconditional loads from clamped addresses + selects.
      const int kAc = okA ? kA : 0;
      const int kBc = okB ? kB : 0;
      const float    av  = Arow[kAc];
      const uint16_t b0r = Br0[kBc];
      const uint16_t b1r = Br1[kBc];
      const uint16_t b2r = Br2[kBc];
      const uint16_t b3r = Br3[kBc];
      a[j]   = okA ? f2bf(av) : bfzero();
      b0v[j] = okB ? __builtin_bit_cast(__bf16, b0r) : bfzero();
      b1v[j] = okB ? __builtin_bit_cast(__bf16, b1r) : bfzero();
      b2v[j] = okB ? __builtin_bit_cast(__bf16, b2r) : bfzero();
      b3v[j] = okB ? __builtin_bit_cast(__bf16, b3r) : bfzero();
    }
    c0 = __builtin_amdgcn_wmma_f32_16x16x32_bf16(false, a, false, b0v, (short)0, c0, false, false);
    c1 = __builtin_amdgcn_wmma_f32_16x16x32_bf16(false, a, false, b1v, (short)0, c1, false, false);
    c2 = __builtin_amdgcn_wmma_f32_16x16x32_bf16(false, a, false, b2v, (short)0, c2, false, false);
    c3 = __builtin_amdgcn_wmma_f32_16x16x32_bf16(false, a, false, b3v, (short)0, c3, false, false);
  }
  // C layout: VGPR r -> M = r (lanes 0-15) / 8+r (lanes 16-31); N = lane%16
  float* outBase = i2h + (size_t)(m0 + laneHi * 8) * H_ + n0 + laneLo;
#pragma unroll
  for (int r = 0; r < 8; ++r) {
    float* o = outBase + (size_t)r * H_;
    o[0]  = c0[r];
    o[16] = c1[r];
    o[32] = c2[r];
    o[48] = c3[r];
  }
}

// ---------------------------------------------------------------------------
// Phase 2: recurrence. One block per 16 batch rows (8 blocks total), 512
// threads = 16 waves. Per step: S[16,512] @ Wh^T via 32 WMMA tiles (2/wave),
// spikes (bf16) + accumulators staged in LDS, LIF update elementwise.
// Output-layer math is dead code w.r.t. the returned tensor and is skipped.
// ---------------------------------------------------------------------------
__global__ __launch_bounds__(512) void recurrent_kernel(
    const float* __restrict__ i2h, const uint16_t* __restrict__ whbf,
    const float* __restrict__ bi, const float* __restrict__ bh,
    const float* __restrict__ thr_h,
    const float* __restrict__ mem0, const float* __restrict__ spk0,
    uint8_t* __restrict__ spk) {
  __shared__ __align__(16) uint16_t spkA[16 * H_];   // 16 KB: spike matrix (bf16)
  __shared__ float accS[16 * H_];                    // 32 KB: GEMM result
  const int tid    = threadIdx.x;
  const int lane   = tid & 31;
  const int wave   = tid >> 5;          // 0..15; also elementwise row index
  const int laneLo = lane & 15;
  const int laneHi = (lane >> 4) & 1;
  const int b0 = blockIdx.x * 16;

  float memv[16], biasv[16], thrv[16];
#pragma unroll
  for (int e = 0; e < 16; ++e) {
    int h = lane + 32 * e;
    memv[e]  = mem0[(size_t)(b0 + wave) * H_ + h];
    biasv[e] = bi[h] + bh[h];
    thrv[e]  = thr_h[h];
    spkA[wave * H_ + h] =
        __builtin_bit_cast(uint16_t, f2bf(spk0[(size_t)(b0 + wave) * H_ + h]));
  }
  __syncthreads();

  // B rows for this wave's two N tiles: n = wave*32 + laneLo (+16)
  const uint16_t* BrA = whbf + (size_t)(wave * 32 + laneLo) * H_;
  const uint16_t* BrB = BrA + (size_t)16 * H_;

  for (int t = 0; t < T_; ++t) {
    v8f c0 = {}, c1 = {};
#pragma unroll
    for (int kk = 0; kk < 16; ++kk) {   // K = 512 in steps of 32
      const int kbA = kk * 32 + laneHi * 16;
      const int kbB = kk * 32 + laneHi * 16;
      const v8bf* arow = reinterpret_cast<const v8bf*>(spkA + (size_t)laneLo * H_);
      v8bf lo = arow[kbA >> 3];
      v8bf hi = arow[(kbA >> 3) + 1];
      v16bf a = __builtin_shufflevector(lo, hi,
          0, 1, 2, 3, 4, 5, 6, 7, 8, 9, 10, 11, 12, 13, 14, 15);
      v16bf bv0 = ld16bf(BrA + kbB);
      v16bf bv1 = ld16bf(BrB + kbB);
      c0 = __builtin_amdgcn_wmma_f32_16x16x32_bf16(false, a, false, bv0, (short)0, c0, false, false);
      c1 = __builtin_amdgcn_wmma_f32_16x16x32_bf16(false, a, false, bv1, (short)0, c1, false, false);
    }
#pragma unroll
    for (int r = 0; r < 8; ++r) {
      accS[(r + laneHi * 8) * H_ + wave * 32 + laneLo]      = c0[r];
      accS[(r + laneHi * 8) * H_ + wave * 32 + 16 + laneLo] = c1[r];
    }
    __syncthreads();   // GEMM reads of spkA done; accS visible

    const float* xr = i2h + ((size_t)(b0 + wave) * T_ + t) * H_;
    uint8_t*     so = spk + ((size_t)(b0 + wave) * T_ + t) * H_;
#pragma unroll
    for (int e = 0; e < 16; ++e) {
      int h = lane + 32 * e;
      // mem_update (DECAY=1): mem += in; spike=(mem-thr>0); reset; clamp at -thr
      float m = memv[e] + accS[wave * H_ + h] + xr[h] + biasv[e];
      float spike = (m - thrv[e] > 0.0f) ? 1.0f : 0.0f;
      m = (spike != 0.0f) ? 0.0f : m;
      if (m < -thrv[e]) m = -thrv[e];
      memv[e] = m;
      so[h] = (uint8_t)spike;
      spkA[wave * H_ + h] = (spike != 0.0f) ? (uint16_t)0x3F80   // bf16(1.0)
                                            : (uint16_t)0;
    }
    __syncthreads();   // new spkA visible before next step's GEMM
  }
}

// ---------------------------------------------------------------------------
// Phase 3: moving average, window 10, stride 1 -> out[b, h, j], j=0..190
// One thread per (b,h); register ring buffer; reads coalesced over h.
// ---------------------------------------------------------------------------
__global__ __launch_bounds__(256) void movavg_kernel(
    const uint8_t* __restrict__ spk, float* __restrict__ out) {
  int i = blockIdx.x * 256 + threadIdx.x;   // 0..65535
  int b = i >> 9, h = i & (H_ - 1);
  const uint8_t* sp = spk + (size_t)b * T_ * H_ + h;
  float* o = out + ((size_t)b * H_ + h) * OUTW;
  float w[WWIN];
#pragma unroll
  for (int j = 0; j < WWIN; ++j) w[j] = 0.0f;
  float s = 0.0f;
  for (int t = 0; t < T_; ++t) {
    float v = (float)sp[(size_t)t * H_];
    s += v - w[0];
#pragma unroll
    for (int j = 0; j < WWIN - 1; ++j) w[j] = w[j + 1];
    w[WWIN - 1] = v;
    if (t >= WWIN - 1) o[t - (WWIN - 1)] = s * 0.1f;
  }
}

// ---------------------------------------------------------------------------
extern "C" void kernel_launch(void* const* d_in, const int* in_sizes, int n_in,
                              void* d_out, int out_size, void* d_ws, size_t ws_size,
                              hipStream_t stream) {
  (void)in_sizes; (void)n_in; (void)out_size; (void)ws_size;
  const float* inp   = (const float*)d_in[0];
  const float* mem0  = (const float*)d_in[2];
  const float* spk0  = (const float*)d_in[3];
  const float* Wi    = (const float*)d_in[6];
  const float* bi    = (const float*)d_in[7];
  const float* Wh    = (const float*)d_in[8];
  const float* bh    = (const float*)d_in[9];
  const float* thr_h = (const float*)d_in[12];
  // d_in[1] labels, d_in[4,5] output state, d_in[10,11] Wo/bo, d_in[13] thr_o:
  // unused — they do not affect the returned tensor.

  char* ws = (char*)d_ws;
  float*    i2h  = (float*)(ws);                 // 52,428,800 B  [B,T,H] fp32
  uint16_t* wibf = (uint16_t*)(ws + 52428800);   //  2,367,488 B  Wi bf16
  uint16_t* whbf = (uint16_t*)(ws + 54796288);   //    524,288 B  Wh bf16
  uint8_t*  spk  = (uint8_t*)(ws + 55320576);    // 13,107,200 B  spikes u8 [B,T,H]

  cvt_bf16_kernel<<<(H_ * IN_ + 255) / 256, 256, 0, stream>>>(Wi, wibf, H_ * IN_);
  cvt_bf16_kernel<<<(H_ * H_ + 255) / 256, 256, 0, stream>>>(Wh, whbf, H_ * H_);
  i2h_gemm_kernel<<<1600, 256, 0, stream>>>(inp, wibf, i2h);
  recurrent_kernel<<<B_ / 16, 512, 0, stream>>>(i2h, whbf, bi, bh, thr_h,
                                                mem0, spk0, spk);
  movavg_kernel<<<(B_ * H_) / 256, 256, 0, stream>>>(spk, (float*)d_out);
}
